// MEGNet_66881230733442
// MI455X (gfx1250) — matrix operand order repaired and checked by reading, not compiled, not run on hardware
//
#include <hip/hip_runtime.h>
#include <math.h>

typedef __attribute__((ext_vector_type(16))) _Float16 v16h;
typedef __attribute__((ext_vector_type(8)))  _Float16 v8h;
typedef __attribute__((ext_vector_type(8)))  float    v8f;

// ---------------------------------------------------------------------------
// WMMA helpers (CDNA5 / gfx1250, wave32, V_WMMA_F32_16X16X32_F16)
// ---------------------------------------------------------------------------
__device__ __forceinline__ v8f zero8() {
  v8f z;
#pragma unroll
  for (int i = 0; i < 8; ++i) z[i] = 0.f;
  return z;
}

__device__ __forceinline__ v16h cat88(v8h lo, v8h hi) {
  return __builtin_shufflevector(lo, hi, 0, 1, 2, 3, 4, 5, 6, 7, 8, 9, 10, 11, 12, 13, 14, 15);
}

__device__ __forceinline__ v8f wmma16(v16h a, v16h b, v8f c) {
  return __builtin_amdgcn_wmma_f32_16x16x32_f16(false, a, false, b, (short)0, c, false, false);
}

// ACT: 0 = none, 1 = relu, 2 = softplus
__device__ __forceinline__ float act_apply(float x, int ACT) {
  if (ACT == 1) return fmaxf(x, 0.f);
  if (ACT == 2) return (x > 20.f) ? x : log1pf(__expf(x));
  return x;
}

__device__ __forceinline__ float sigf(float x) { return 1.f / (1.f + __expf(-x)); }

// A fragment (16 rows x 32 k) from fp32 row-major X, with K bound guard.
// ISA layout: lanes 0-15 hold row M=lane, K = {k0..k0+7, k0+16..k0+23};
//             lanes 16-31 hold K = {k0+8..k0+15, k0+24..k0+31}.
__device__ __forceinline__ v16h ldA_g32(const float* X, int ldx, int row0, int k0, int K, int lane) {
  const float* p = X + (size_t)(row0 + (lane & 15)) * ldx;
  int kb = k0 + ((lane >> 4) << 3);
  v16h a;
#pragma unroll
  for (int j = 0; j < 8; ++j) {
    int k1 = kb + j, k2 = kb + 16 + j;
    a[j]     = (_Float16)((k1 < K) ? p[k1] : 0.f);
    a[j + 8] = (_Float16)((k2 < K) ? p[k2] : 0.f);
  }
  return a;
}

// Unguarded f32 A fragment: 'prow' is this lane's row base pointer.
__device__ __forceinline__ v16h ldA_g32_nc(const float* prow, int koff, int lane) {
  const float* p = prow + koff + ((lane >> 4) << 3);
  v16h a;
#pragma unroll
  for (int j = 0; j < 8; ++j) {
    a[j]     = (_Float16)p[j];
    a[j + 8] = (_Float16)p[16 + j];
  }
  return a;
}

// f16 A fragment straight from memory (global or LDS): two 16-byte loads.
__device__ __forceinline__ v16h ldA_h16(const _Float16* X, int ldx, int row0, int k0, int lane) {
  const _Float16* p = X + (size_t)(row0 + (lane & 15)) * ldx + k0 + ((lane >> 4) << 3);
  v8h lo = *(const v8h*)p;
  v8h hi = *(const v8h*)(p + 16);
  return cat88(lo, hi);
}

// f16 A fragment from a lane's row base pointer.
__device__ __forceinline__ v16h ldA_h16_nc(const _Float16* prow, int koff, int lane) {
  const _Float16* p = prow + koff + ((lane >> 4) << 3);
  v8h lo = *(const v8h*)p;
  v8h hi = *(const v8h*)(p + 16);
  return cat88(lo, hi);
}

// B fragment (Kx16 = W^T tile) from f16 LDS weights, row-major W [NOUTxKP].
__device__ __forceinline__ v16h ldB_lds(const _Float16* Wl, int KP, int n0, int k0, int lane) {
  const _Float16* p = Wl + (size_t)(n0 + (lane & 15)) * KP + k0 + ((lane >> 4) << 3);
  v8h lo = *(const v8h*)p;
  v8h hi = *(const v8h*)(p + 16);
  return cat88(lo, hi);
}

// Store a 16x16 f32 C tile into f16 LDS staging with bias + activation.
// C layout: element r <-> row M = 8*(lane>=16)+r, col N = lane&15.
__device__ __forceinline__ void stC_lds(_Float16* S, int lds, int n0, v8f c,
                                        const float* bl, int ACT, int lane) {
  int col = n0 + (lane & 15);
  int rb = (lane >> 4) << 3;
  float bias = bl[col];
#pragma unroll
  for (int r = 0; r < 8; ++r)
    S[(size_t)(rb + r) * lds + col] = (_Float16)act_apply(c[r] + bias, ACT);
}

// order-preserving float <-> uint mapping for atomicMax on floats
__device__ __forceinline__ unsigned fmap(float x) {
  unsigned u = __float_as_uint(x);
  return (u & 0x80000000u) ? ~u : (u | 0x80000000u);
}
__device__ __forceinline__ float funmap(unsigned u) {
  return __uint_as_float((u & 0x80000000u) ? (u & 0x7FFFFFFFu) : ~u);
}

// ---------------------------------------------------------------------------
// Generic single-layer row GEMM: Y[M x NOUT] (+)= act(X[M x K] @ W^T + b)
// X fp32. W global fp32 [NOUT x K] staged to LDS f16 (zero-padded to KP).
// OF16 = 1 stores Y as f16 (no ACC). Block = 256 threads = 8 waves.
// grid.y indexes groups of NT n-tiles (NT*16 columns each).
// ---------------------------------------------------------------------------
template <int K, int NOUT, int NT, int ACT, int ACC, int OF16>
__global__ __launch_bounds__(256) void gemm_rows(const float* __restrict__ X, int ldx,
                                                 const float* __restrict__ W,
                                                 const float* __restrict__ bias,
                                                 void* __restrict__ Yv, int ldy, int M) {
  constexpr int KP = (K + 31) & ~31;
  __shared__ alignas(16) _Float16 Wl[NT * 16 * KP];
  __shared__ float bl[NT * 16];
  const int tid = threadIdx.x, w = tid >> 5, lane = tid & 31;
  const int nbase = blockIdx.y * NT * 16;
  for (int i = tid; i < NT * 16 * KP; i += 256) {
    int n = i / KP, k = i - n * KP;
    Wl[i] = (k < K) ? (_Float16)W[(size_t)(nbase + n) * K + k] : (_Float16)0.f;
  }
  if (tid < NT * 16) bl[tid] = bias ? bias[nbase + tid] : 0.f;
  __syncthreads();

  int row0 = (blockIdx.x * 8 + w) * 16;
  if (row0 >= M) return;

  v8f acc[NT];
#pragma unroll
  for (int nt = 0; nt < NT; ++nt) acc[nt] = zero8();
#pragma unroll
  for (int k0 = 0; k0 < KP; k0 += 32) {
    v16h a = ldA_g32(X, ldx, row0, k0, K, lane);
#pragma unroll
    for (int nt = 0; nt < NT; ++nt)
      acc[nt] = wmma16(a, ldB_lds(Wl, KP, nt * 16, k0, lane), acc[nt]);
  }
  const int colb = lane & 15, rb = (lane >> 4) << 3;
#pragma unroll
  for (int nt = 0; nt < NT; ++nt) {
    int col = nbase + nt * 16 + colb;
    float b = bl[nt * 16 + colb];
#pragma unroll
    for (int r = 0; r < 8; ++r) {
      size_t idx = (size_t)(row0 + rb + r) * ldy + col;
      float v = acc[nt][r] + b;
      if constexpr (OF16) {
        ((_Float16*)Yv)[idx] = (_Float16)act_apply(v, ACT);
      } else {
        float* Y = (float*)Yv;
        if (ACC) v += Y[idx];
        Y[idx] = act_apply(v, ACT);
      }
    }
  }
}

// ---------------------------------------------------------------------------
// Fused 2-layer dense: X[M x 64] f16 -> act(64->128) -> (128->64) -> X in place
// Block = 128 threads = 4 waves; one 16-row tile per wave.
// ---------------------------------------------------------------------------
template <int ACT_MID>
__global__ __launch_bounds__(128) void mlp2_kernel(_Float16* __restrict__ X,
                                                   const float* __restrict__ W0,
                                                   const float* __restrict__ b0,
                                                   const float* __restrict__ W1,
                                                   const float* __restrict__ b1, int M) {
  __shared__ alignas(16) _Float16 W0l[128 * 64];
  __shared__ alignas(16) _Float16 W1l[64 * 128];
  __shared__ alignas(16) _Float16 stage[4][16 * 128];
  __shared__ float b0l[128], b1l[64];
  const int tid = threadIdx.x, w = tid >> 5, lane = tid & 31;
  for (int i = tid; i < 128 * 64; i += 128) W0l[i] = (_Float16)W0[i];
  for (int i = tid; i < 64 * 128; i += 128) W1l[i] = (_Float16)W1[i];
  if (tid < 128) b0l[tid] = b0[tid];
  if (tid < 64) b1l[tid] = b1[tid];

  int tile = blockIdx.x * 4 + w;
  bool active = (tile * 16) < M;
  int row0 = active ? tile * 16 : 0;
  __syncthreads();

  // layer 0: 64 -> 128
  v8f acc[8];
#pragma unroll
  for (int nt = 0; nt < 8; ++nt) acc[nt] = zero8();
#pragma unroll
  for (int k0 = 0; k0 < 64; k0 += 32) {
    v16h a = ldA_h16(X, 64, row0, k0, lane);
#pragma unroll
    for (int nt = 0; nt < 8; ++nt)
      acc[nt] = wmma16(a, ldB_lds(W0l, 64, nt * 16, k0, lane), acc[nt]);
  }
#pragma unroll
  for (int nt = 0; nt < 8; ++nt) stC_lds(stage[w], 128, nt * 16, acc[nt], b0l, ACT_MID, lane);
  __syncthreads();

  // layer 1: 128 -> 64
  v8f acc2[4];
#pragma unroll
  for (int nt = 0; nt < 4; ++nt) acc2[nt] = zero8();
#pragma unroll
  for (int k0 = 0; k0 < 128; k0 += 32) {
    v16h a = ldA_h16(stage[w], 128, 0, k0, lane);
#pragma unroll
    for (int nt = 0; nt < 4; ++nt)
      acc2[nt] = wmma16(a, ldB_lds(W1l, 128, nt * 16, k0, lane), acc2[nt]);
  }
  if (active) {
    const int colb = lane & 15, rb = (lane >> 4) << 3;
#pragma unroll
    for (int nt = 0; nt < 4; ++nt) {
      int col = nt * 16 + colb;
      float b = b1l[col];
#pragma unroll
      for (int r = 0; r < 8; ++r)
        X[(size_t)(row0 + rb + r) * 64 + col] = (_Float16)(acc2[nt][r] + b);
    }
  }
}

// ---------------------------------------------------------------------------
// Fused edge update: ec = [h_n[src], h_n[dst], h_e] (192) -> softplus MLP
// 192->64->64->64, writes h_e (f16) in place, scatter-adds f32 into agg[dst].
// Block = 256 threads = 8 waves; one 16-edge tile per wave.
// ---------------------------------------------------------------------------
__global__ __launch_bounds__(256) void edge_block_kernel(
    const _Float16* __restrict__ h_n, _Float16* __restrict__ h_e,
    const int* __restrict__ src, const int* __restrict__ dst,
    const float* __restrict__ W0, const float* __restrict__ b0,
    const float* __restrict__ W1, const float* __restrict__ b1,
    const float* __restrict__ W2, const float* __restrict__ b2,
    float* __restrict__ agg, int E_) {
  __shared__ alignas(16) _Float16 W0l[64 * 192];
  __shared__ alignas(16) _Float16 W1l[64 * 64];
  __shared__ alignas(16) _Float16 W2l[64 * 64];
  __shared__ alignas(16) _Float16 stage[8][16 * 64];
  __shared__ float b0l[64], b1l[64], b2l[64];
  __shared__ int dstl[8][16];
  const int tid = threadIdx.x, w = tid >> 5, lane = tid & 31;
  for (int i = tid; i < 64 * 192; i += 256) W0l[i] = (_Float16)W0[i];
  for (int i = tid; i < 64 * 64; i += 256) {
    W1l[i] = (_Float16)W1[i];
    W2l[i] = (_Float16)W2[i];
  }
  if (tid < 64) { b0l[tid] = b0[tid]; b1l[tid] = b1[tid]; b2l[tid] = b2[tid]; }

  int tile = blockIdx.x * 8 + w;
  bool active = (tile * 16) < E_;
  int row0 = active ? tile * 16 : 0;
  int myrow = row0 + (lane & 15);
  int sIdx = src[myrow], dIdx = dst[myrow];
  if (lane < 16) dstl[w][lane] = dst[row0 + lane];
  __syncthreads();

  // layer 0: K = 192 (gathered concat), softplus
  const _Float16* srcp = h_n + (size_t)sIdx * 64;
  const _Float16* dstp = h_n + (size_t)dIdx * 64;
  const _Float16* ep = h_e + (size_t)myrow * 64;
  v8f acc[4];
#pragma unroll
  for (int nt = 0; nt < 4; ++nt) acc[nt] = zero8();
#pragma unroll
  for (int ks = 0; ks < 6; ++ks) {
    const int k0 = ks * 32;
    const _Float16* base = (k0 < 64) ? srcp : (k0 < 128) ? dstp : ep;
    v16h a = ldA_h16_nc(base, k0 & 63, lane);
#pragma unroll
    for (int nt = 0; nt < 4; ++nt)
      acc[nt] = wmma16(a, ldB_lds(W0l, 192, nt * 16, k0, lane), acc[nt]);
  }
#pragma unroll
  for (int nt = 0; nt < 4; ++nt) stC_lds(stage[w], 64, nt * 16, acc[nt], b0l, 2, lane);
  __syncthreads();

  // layer 1: 64 -> 64, softplus
  v8f acc2[4];
#pragma unroll
  for (int nt = 0; nt < 4; ++nt) acc2[nt] = zero8();
#pragma unroll
  for (int k0 = 0; k0 < 64; k0 += 32) {
    v16h a = ldA_h16(stage[w], 64, 0, k0, lane);
#pragma unroll
    for (int nt = 0; nt < 4; ++nt)
      acc2[nt] = wmma16(a, ldB_lds(W1l, 64, nt * 16, k0, lane), acc2[nt]);
  }
  __syncthreads();  // all stage reads done before overwrite
#pragma unroll
  for (int nt = 0; nt < 4; ++nt) stC_lds(stage[w], 64, nt * 16, acc2[nt], b1l, 2, lane);
  __syncthreads();

  // layer 2: 64 -> 64, linear
  v8f acc3[4];
#pragma unroll
  for (int nt = 0; nt < 4; ++nt) acc3[nt] = zero8();
#pragma unroll
  for (int k0 = 0; k0 < 64; k0 += 32) {
    v16h a = ldA_h16(stage[w], 64, 0, k0, lane);
#pragma unroll
    for (int nt = 0; nt < 4; ++nt)
      acc3[nt] = wmma16(a, ldB_lds(W2l, 64, nt * 16, k0, lane), acc3[nt]);
  }
  if (active) {
    const int colb = lane & 15, rb = (lane >> 4) << 3;
#pragma unroll
    for (int nt = 0; nt < 4; ++nt) {
      int col = nt * 16 + colb;
      float b = b2l[col];
#pragma unroll
      for (int r = 0; r < 8; ++r) {
        float v = acc3[nt][r] + b;
        h_e[(size_t)(row0 + rb + r) * 64 + col] = (_Float16)v;
        atomicAdd(&agg[(size_t)dstl[w][rb + r] * 64 + col], v);
      }
    }
  }
}

// ---------------------------------------------------------------------------
// Fused node update: nc = [h_n (f16), agg (f32)] (128) -> 128->64->64->64
// softplus MLP; writes h_n (f16) in place. Block = 256 threads = 8 waves.
// ---------------------------------------------------------------------------
__global__ __launch_bounds__(256) void node_block_kernel(
    _Float16* __restrict__ h_n, const float* __restrict__ agg,
    const float* __restrict__ W0, const float* __restrict__ b0,
    const float* __restrict__ W1, const float* __restrict__ b1,
    const float* __restrict__ W2, const float* __restrict__ b2, int N_) {
  __shared__ alignas(16) _Float16 W0l[64 * 128];
  __shared__ alignas(16) _Float16 W1l[64 * 64];
  __shared__ alignas(16) _Float16 W2l[64 * 64];
  __shared__ alignas(16) _Float16 stage[8][16 * 64];
  __shared__ float b0l[64], b1l[64], b2l[64];
  const int tid = threadIdx.x, w = tid >> 5, lane = tid & 31;
  for (int i = tid; i < 64 * 128; i += 256) W0l[i] = (_Float16)W0[i];
  for (int i = tid; i < 64 * 64; i += 256) {
    W1l[i] = (_Float16)W1[i];
    W2l[i] = (_Float16)W2[i];
  }
  if (tid < 64) { b0l[tid] = b0[tid]; b1l[tid] = b1[tid]; b2l[tid] = b2[tid]; }

  int tile = blockIdx.x * 8 + w;
  bool active = (tile * 16) < N_;
  int row0 = active ? tile * 16 : 0;
  int myrow = row0 + (lane & 15);
  __syncthreads();

  // layer 0: K = 128 ([h_n | agg]), softplus
  const _Float16* np = h_n + (size_t)myrow * 64;
  const float* ap = agg + (size_t)myrow * 64;
  v8f acc[4];
#pragma unroll
  for (int nt = 0; nt < 4; ++nt) acc[nt] = zero8();
#pragma unroll
  for (int ks = 0; ks < 4; ++ks) {
    const int k0 = ks * 32;
    v16h a = (k0 < 64) ? ldA_h16_nc(np, k0, lane) : ldA_g32_nc(ap, k0 & 63, lane);
#pragma unroll
    for (int nt = 0; nt < 4; ++nt)
      acc[nt] = wmma16(a, ldB_lds(W0l, 128, nt * 16, k0, lane), acc[nt]);
  }
#pragma unroll
  for (int nt = 0; nt < 4; ++nt) stC_lds(stage[w], 64, nt * 16, acc[nt], b0l, 2, lane);
  __syncthreads();

  v8f acc2[4];
#pragma unroll
  for (int nt = 0; nt < 4; ++nt) acc2[nt] = zero8();
#pragma unroll
  for (int k0 = 0; k0 < 64; k0 += 32) {
    v16h a = ldA_h16(stage[w], 64, 0, k0, lane);
#pragma unroll
    for (int nt = 0; nt < 4; ++nt)
      acc2[nt] = wmma16(a, ldB_lds(W1l, 64, nt * 16, k0, lane), acc2[nt]);
  }
  __syncthreads();
#pragma unroll
  for (int nt = 0; nt < 4; ++nt) stC_lds(stage[w], 64, nt * 16, acc2[nt], b1l, 2, lane);
  __syncthreads();

  v8f acc3[4];
#pragma unroll
  for (int nt = 0; nt < 4; ++nt) acc3[nt] = zero8();
#pragma unroll
  for (int k0 = 0; k0 < 64; k0 += 32) {
    v16h a = ldA_h16(stage[w], 64, 0, k0, lane);
#pragma unroll
    for (int nt = 0; nt < 4; ++nt)
      acc3[nt] = wmma16(a, ldB_lds(W2l, 64, nt * 16, k0, lane), acc3[nt]);
  }
  if (active) {
    const int colb = lane & 15, rb = (lane >> 4) << 3;
#pragma unroll
    for (int nt = 0; nt < 4; ++nt) {
      int col = nt * 16 + colb;
      float b = b2l[col];
#pragma unroll
      for (int r = 0; r < 8; ++r)
        h_n[(size_t)(row0 + rb + r) * 64 + col] = (_Float16)(acc3[nt][r] + b);
    }
  }
}

// ---------------------------------------------------------------------------
// Set2Set / misc elementwise kernels
// ---------------------------------------------------------------------------
__global__ void seg_from_src_kernel(const int* __restrict__ src, const int* __restrict__ batch,
                                    int* __restrict__ seg_e, int E_) {
  int i = blockIdx.x * blockDim.x + threadIdx.x;
  if (i < E_) seg_e[i] = batch[src[i]];
}

__global__ void lstm_cell_kernel(const float* __restrict__ gates, float* __restrict__ h,
                                 float* __restrict__ c, float* __restrict__ q /* ld 256 */) {
  int gid = blockIdx.x * blockDim.x + threadIdx.x;
  if (gid >= 512 * 64) return;
  int b = gid >> 6, u = gid & 63;
  const float* g = gates + (size_t)b * 256;
  float gi = g[u], gf = g[64 + u], gg = g[128 + u], go = g[192 + u];
  float cv = sigf(gf) * c[gid] + sigf(gi) * tanhf(gg);
  float hv = sigf(go) * tanhf(cv);
  c[gid] = cv;
  h[gid] = hv;
  q[(size_t)b * 256 + u] = hv;
}

__global__ void attn_init_kernel(float* __restrict__ rreg /* ld 256 */, unsigned* __restrict__ m,
                                 float* __restrict__ denom) {
  int gid = blockIdx.x * blockDim.x + threadIdx.x;
  if (gid < 512 * 64) rreg[(size_t)(gid >> 6) * 256 + (gid & 63)] = 0.f;
  if (gid < 512) { m[gid] = 0u; denom[gid] = 0.f; }
}

__global__ void attn_dot_max_kernel(const _Float16* __restrict__ xf, const int* __restrict__ seg,
                                    const float* __restrict__ q /* ld 256 */,
                                    float* __restrict__ ebuf, unsigned* __restrict__ m, int M) {
  int i = blockIdx.x * blockDim.x + threadIdx.x;
  if (i >= M) return;
  int s = seg[i];
  const _Float16* xp = xf + (size_t)i * 64;
  const float* qp = q + (size_t)s * 256;
  float dot = 0.f;
#pragma unroll 16
  for (int k = 0; k < 64; ++k) dot += (float)xp[k] * qp[k];
  ebuf[i] = dot;
  atomicMax(&m[s], fmap(dot));
}

__global__ void attn_expsum_kernel(float* __restrict__ ebuf, const int* __restrict__ seg,
                                   const unsigned* __restrict__ m, float* __restrict__ denom,
                                   int M) {
  int i = blockIdx.x * blockDim.x + threadIdx.x;
  if (i >= M) return;
  int s = seg[i];
  float a = __expf(ebuf[i] - funmap(m[s]));
  ebuf[i] = a;
  atomicAdd(&denom[s], a);
}

__global__ void attn_scatter_kernel(const _Float16* __restrict__ xf, const float* __restrict__ a,
                                    const int* __restrict__ seg, const float* __restrict__ denom,
                                    float* __restrict__ r /* ld 256 */, int M) {
  long long gid = (long long)blockIdx.x * blockDim.x + threadIdx.x;
  if (gid >= (long long)M * 64) return;
  int i = (int)(gid >> 6), k = (int)(gid & 63);
  int s = seg[i];
  float w = a[i] / denom[s];
  atomicAdd(&r[(size_t)s * 256 + k], w * (float)xf[(size_t)i * 64 + k]);
}

__global__ void out_kernel(const float* __restrict__ X /* 512x256 */,
                           const float* __restrict__ W, const float* __restrict__ b,
                           float* __restrict__ y) {
  int i = blockIdx.x * blockDim.x + threadIdx.x;
  if (i >= 512) return;
  const float* xp = X + (size_t)i * 256;
  float s = 0.f;
#pragma unroll 16
  for (int k = 0; k < 256; ++k) s += xp[k] * W[k];
  y[i] = s + b[0];
}

// ---------------------------------------------------------------------------
// Host-side orchestration
// ---------------------------------------------------------------------------
extern "C" void kernel_launch(void* const* d_in, const int* in_sizes, int n_in,
                              void* d_out, int out_size, void* d_ws, size_t ws_size,
                              hipStream_t stream) {
  const int N = in_sizes[0] / 32;   // nodes
  const int E = in_sizes[2] / 16;   // edges
  const float* x = (const float*)d_in[0];
  const int* ei = (const int*)d_in[1];
  const float* ea = (const float*)d_in[2];
  const int* batch = (const int*)d_in[3];
  const int* src = ei;
  const int* dst = ei + E;
  auto P = [&](int i) { return (const float*)d_in[i]; };

  // workspace bump allocator
  char* ws = (char*)d_ws;
  size_t off = 0;
  auto alloc = [&](size_t bytes) -> void* {
    off = (off + 255) & ~(size_t)255;
    void* p = ws + off;
    off += bytes;
    return p;
  };
  _Float16* h_n = (_Float16*)alloc((size_t)N * 64 * 2);
  _Float16* h_e = (_Float16*)alloc((size_t)E * 64 * 2);
  float* agg    = (float*)alloc((size_t)N * 64 * 4);
  float* ebuf   = (float*)alloc((size_t)E * 4);
  int* seg_e    = (int*)alloc((size_t)E * 4);
  float* gates  = (float*)alloc(512 * 256 * 4);
  float* hb     = (float*)alloc(512 * 64 * 4);
  float* cb     = (float*)alloc(512 * 64 * 4);
  float* cat    = (float*)alloc(512 * 256 * 4);
  unsigned* mv  = (unsigned*)alloc(512 * 4);
  float* denom  = (float*)alloc(512 * 4);
  float* t1     = (float*)alloc(512 * 512 * 4);
  float* t2     = (float*)alloc(512 * 256 * 4);

  // embeddings (f32 inputs -> f16 activations)
  gemm_rows<32, 64, 4, 0, 0, 1><<<dim3((N + 127) / 128, 1), 256, 0, stream>>>(x, 32, P(4), P(5), h_n, 64, N);
  gemm_rows<16, 64, 4, 0, 0, 1><<<dim3((E + 127) / 128, 1), 256, 0, stream>>>(ea, 16, P(6), P(7), h_e, 64, E);

  for (int d = 0; d < 2; ++d) {
    const int dn = 8 + 4 * d;    // dense_nodes[d]: W0,b0,W1,b1
    const int de = 16 + 4 * d;   // dense_edges[d]
    const int em = 24 + 12 * d;  // blocks[d].edge_mlp: W0,b0,W1,b1,W2,b2
    const int nm = em + 6;       // blocks[d].node_mlp
    mlp2_kernel<1><<<(N + 63) / 64, 128, 0, stream>>>(h_n, P(dn), P(dn + 1), P(dn + 2), P(dn + 3), N);
    mlp2_kernel<0><<<(E + 63) / 64, 128, 0, stream>>>(h_e, P(de), P(de + 1), P(de + 2), P(de + 3), E);
    hipMemsetAsync(agg, 0, (size_t)N * 64 * 4, stream);
    edge_block_kernel<<<(E + 127) / 128, 256, 0, stream>>>(h_n, h_e, src, dst,
        P(em), P(em + 1), P(em + 2), P(em + 3), P(em + 4), P(em + 5), agg, E);
    node_block_kernel<<<(N + 127) / 128, 256, 0, stream>>>(h_n, agg,
        P(nm), P(nm + 1), P(nm + 2), P(nm + 3), P(nm + 4), P(nm + 5), N);
  }

  // Set2Set pooling (nodes -> cat[:,0:128], edges -> cat[:,128:256])
  seg_from_src_kernel<<<(E + 255) / 256, 256, 0, stream>>>(src, batch, seg_e, E);
  hipMemsetAsync(cat, 0, 512 * 256 * 4, stream);

  auto run_s2s = [&](const _Float16* xf, const int* seg, int M, int coloff, int pw) {
    hipMemsetAsync(hb, 0, 512 * 64 * 4, stream);
    hipMemsetAsync(cb, 0, 512 * 64 * 4, stream);
    for (int step = 0; step < 3; ++step) {
      // gates = q_star @ W_ih^T + b_ih + h @ W_hh^T + b_hh
      gemm_rows<128, 256, 4, 0, 0, 0><<<dim3(4, 4), 256, 0, stream>>>(cat + coloff, 256, P(pw), P(pw + 2), gates, 256, 512);
      gemm_rows<64, 256, 4, 0, 1, 0><<<dim3(4, 4), 256, 0, stream>>>(hb, 64, P(pw + 1), P(pw + 3), gates, 256, 512);
      lstm_cell_kernel<<<128, 256, 0, stream>>>(gates, hb, cb, cat + coloff);
      attn_init_kernel<<<(512 * 64 + 255) / 256, 256, 0, stream>>>(cat + coloff + 64, mv, denom);
      int blk = (M + 255) / 256;
      attn_dot_max_kernel<<<blk, 256, 0, stream>>>(xf, seg, cat + coloff, ebuf, mv, M);
      attn_expsum_kernel<<<blk, 256, 0, stream>>>(ebuf, seg, mv, denom, M);
      long long tot = (long long)M * 64;
      attn_scatter_kernel<<<(unsigned)((tot + 255) / 256), 256, 0, stream>>>(xf, ebuf, seg, denom, cat + coloff + 64, M);
    }
  };
  run_s2s(h_n, batch, N, 0, 48);    // s2s_nodes: W_ih=48, W_hh=49, b_ih=50, b_hh=51
  run_s2s(h_e, seg_e, E, 128, 52);  // s2s_edges

  // final MLP: 256 -> 512 (relu) -> 256 (relu) -> 1
  gemm_rows<256, 512, 4, 1, 0, 0><<<dim3(4, 8), 256, 0, stream>>>(cat, 256, P(56), P(57), t1, 512, 512);
  gemm_rows<512, 256, 2, 1, 0, 0><<<dim3(4, 8), 256, 0, stream>>>(t1, 512, P(58), P(59), t2, 256, 512);
  out_kernel<<<2, 256, 0, stream>>>(t2, P(60), P(61), (float*)d_out);

  (void)n_in; (void)out_size; (void)ws_size;
}
